// SSTSSM_48052094108261
// MI455X (gfx1250) — matrix operand order repaired and checked by reading, not compiled, not run on hardware
//
#include <hip/hip_runtime.h>
#include <hip/hip_bf16.h>

typedef __attribute__((ext_vector_type(16))) __bf16       v16bf;
typedef __attribute__((ext_vector_type(8)))  float        v8f;
typedef __attribute__((ext_vector_type(8)))  unsigned int v8u;

#define B_     64
#define S_     2048
#define DMODEL 512
#define DINNER 1024
#define DSTATE 16
#define XZS    36   // padded row stride for xz: [0]=dt_raw, [4..19]=B, [20..35]=C

// ---------------------------------------------------------------------------
// helpers
// ---------------------------------------------------------------------------
__device__ __forceinline__ unsigned int pk_bf16(float a, float b) {
  unsigned short ha = __builtin_bit_cast(unsigned short, (__bf16)a);
  unsigned short hb = __builtin_bit_cast(unsigned short, (__bf16)b);
  return (unsigned int)ha | ((unsigned int)hb << 16);
}

// A-matrix 16x32 bf16 frag (ISA 7.12.2 layout): lane<16 -> K {0..7,16..23},
// lane>=16 -> K {8..15,24..31}; row M = lane%16. rowU = 16 dwords (32 halves).
__device__ __forceinline__ v16bf frag_a(const unsigned int* rowU, int lane) {
  const int klo = (lane & 16) ? 4 : 0;   // dword offset (8 halves)
  v8u r;
#pragma unroll
  for (int j = 0; j < 8; ++j) r[j] = rowU[klo + j + ((j >= 4) ? 4 : 0)];
  return __builtin_bit_cast(v16bf, r);
}

// B-matrix 32x16 bf16 frag: lanes 0-15 K {0..7,16..23}, lanes 16-31 K {8..15,24..31}, N=lane%16.
// LDS stores B transposed as [n][k] so pairs are contiguous.
__device__ __forceinline__ v16bf frag_b(const unsigned int* rowU, int lane) {
  const int kb = (lane & 16) ? 4 : 0;
  v8u r;
#pragma unroll
  for (int j = 0; j < 8; ++j) r[j] = rowU[kb + j + ((j >= 4) ? 4 : 0)];
  return __builtin_bit_cast(v16bf, r);
}

// ---------------------------------------------------------------------------
// Kernel 0: fp32 -> bf16 pre-conversion (one shot; lets the GEMM stage tiles
// with raw-byte async copies instead of VALU conversion in the hot loop).
// ---------------------------------------------------------------------------
__global__ __launch_bounds__(256) void cvt_bf16_kernel(const float* __restrict__ src,
                                                       unsigned int* __restrict__ dst,
                                                       int n4) {
  const int i = blockIdx.x * 256 + threadIdx.x;
  if (i >= n4) return;
  const float4 f = ((const float4*)src)[i];
  dst[2 * i]     = pk_bf16(f.x, f.y);
  dst[2 * i + 1] = pk_bf16(f.z, f.w);
}

// ---------------------------------------------------------------------------
// Kernel 1: fused gather + WMMA GEMM:  u[m,n] = sum_k emb[m,k]*W_in[n,k]
// Tile 128(M)x128(N)x32(K); 8 waves, each wave = 64x32 C patch (4x2 WMMA).
// Tiles staged via GLOBAL_LOAD_ASYNC_TO_LDS_B128 with double-buffered LDS.
// ---------------------------------------------------------------------------
__global__ __launch_bounds__(256) void gemm1_kernel(
    const int* __restrict__ x, const __bf16* __restrict__ temb16,
    const __bf16* __restrict__ win16, float* __restrict__ u) {
  // 2 buffers x (256 rows x 32 bf16) = 32 KB.  rows 0..127 = A, 128..255 = B.
  __shared__ __align__(16) __bf16 tile[2 * 256 * 32];

  const int tid   = threadIdx.x;
  const int lane  = tid & 31;
  const int w     = tid >> 5;
  const int wm    = w & 1;    // wave M slot (0..1)
  const int wn    = w >> 1;   // wave N slot (0..3)
  const int mtile = blockIdx.x;   // 0..1023
  const int ntile = blockIdx.y;   // 0..7

  // each thread owns one tile row (gathered embedding row or W_in row)
  const __bf16* rowp;
  if (tid < 128) {                       // waves 0..3: wave-uniform branch
    const int tok = x[(size_t)mtile * 128 + tid];
    rowp = temb16 + (size_t)tok * DMODEL;
  } else {                               // waves 4..7
    rowp = win16 + (size_t)(ntile * 128 + (tid - 128)) * DMODEL;
  }

  const unsigned ldsbase = (unsigned)(unsigned long long)(uintptr_t)&tile[0];

  // issue 4x async b128 = this thread's 64-byte row slab for K-step kk
  auto stage = [&](int bufi, int kk) {
    const unsigned lbase = ldsbase + (unsigned)bufi * 16384u + (unsigned)tid * 64u;
    const unsigned long long ga = (unsigned long long)(uintptr_t)(rowp + kk);
#pragma unroll
    for (int j = 0; j < 4; ++j)
      asm volatile("global_load_async_to_lds_b128 %0, %1, off"
                   :: "v"(lbase + (unsigned)(j * 16)),
                      "v"(ga + (unsigned long long)(j * 16))
                   : "memory");
  };

  v8f acc[4][2];
#pragma unroll
  for (int mi = 0; mi < 4; ++mi)
#pragma unroll
    for (int ni = 0; ni < 2; ++ni) acc[mi][ni] = (v8f)0.0f;

  stage(0, 0);
  int buf = 0;
  for (int k0 = 0; k0 < DMODEL; k0 += 32) {
    asm volatile("s_wait_asynccnt 0x0" ::: "memory");   // my tile writes done
    __syncthreads();                                    // everyone's done + prev reads done
    if (k0 + 32 < DMODEL) stage(buf ^ 1, k0 + 32);      // prefetch next slab

    const unsigned int* AsU = (const unsigned int*)(tile + buf * 8192);
    const unsigned int* BsU = AsU + 128 * 16;

    v16bf af[4], bfr[2];
#pragma unroll
    for (int mi = 0; mi < 4; ++mi)
      af[mi] = frag_a(AsU + (wm * 64 + mi * 16 + (lane & 15)) * 16, lane);
#pragma unroll
    for (int ni = 0; ni < 2; ++ni)
      bfr[ni] = frag_b(BsU + (wn * 32 + ni * 16 + (lane & 15)) * 16, lane);
#pragma unroll
    for (int mi = 0; mi < 4; ++mi)
#pragma unroll
      for (int ni = 0; ni < 2; ++ni)
        acc[mi][ni] = __builtin_amdgcn_wmma_f32_16x16x32_bf16(
            false, af[mi], false, bfr[ni], (short)0, acc[mi][ni], false, false);
    buf ^= 1;
  }

  // C layout: vgpr r, lanes0-15 -> M=r, lanes16-31 -> M=r+8, N=lane%16
  const int    rlane = lane & 15;
  const int    hi    = (lane >> 4) * 8;
  const size_t mbase = (size_t)mtile * 128 + wm * 64;
  const int    nbase = ntile * 128 + wn * 32;
#pragma unroll
  for (int mi = 0; mi < 4; ++mi)
#pragma unroll
    for (int ni = 0; ni < 2; ++ni)
#pragma unroll
      for (int r = 0; r < 8; ++r)
        u[(mbase + mi * 16 + r + hi) * (size_t)DINNER + nbase + ni * 16 + rlane] =
            acc[mi][ni][r];
}

// ---------------------------------------------------------------------------
// Kernel 2: xz[m, j] = sum_i u[m,i] * W_x[j,i]   (j=0..32)  — one wave per row,
// u read exactly once; W_x chunks staged in LDS.
// ---------------------------------------------------------------------------
__global__ __launch_bounds__(256) void xz_kernel(const float* __restrict__ u,
                                                 const float* __restrict__ W_x,
                                                 float* __restrict__ xz) {
  __shared__ float WxS[33 * 128];
  const int tid  = threadIdx.x;
  const int lane = tid & 31;
  const int w    = tid >> 5;
  const size_t m = (size_t)blockIdx.x * 8 + w;
  const float* urow = u + m * (size_t)DINNER;

  float acc[33];
#pragma unroll
  for (int j = 0; j < 33; ++j) acc[j] = 0.0f;

  for (int i0 = 0; i0 < DINNER; i0 += 128) {
    __syncthreads();
    for (int t = tid; t < 33 * 128; t += 256) {
      const int j = t >> 7, c = t & 127;
      WxS[t] = W_x[(size_t)j * DINNER + i0 + c];
    }
    __syncthreads();
#pragma unroll
    for (int ii = 0; ii < 128; ii += 32) {
      const float val = urow[i0 + ii + lane];
#pragma unroll
      for (int j = 0; j < 33; ++j)
        acc[j] = fmaf(val, WxS[j * 128 + ii + lane], acc[j]);
    }
  }
#pragma unroll
  for (int j = 0; j < 33; ++j) {
    float v = acc[j];
    for (int off = 16; off > 0; off >>= 1) v += __shfl_xor(v, off);
    if (lane == 0) xz[m * XZS + (j == 0 ? 0 : j + 3)] = v;
  }
}

// ---------------------------------------------------------------------------
// Kernel 3: selective scan, one thread per (b, d) channel; accumulates the
// valid-masked sum of y over s (pooling is linear, so no ys materialized).
// ---------------------------------------------------------------------------
__global__ __launch_bounds__(256) void scan_kernel(
    const int* __restrict__ x, const float* __restrict__ u,
    const float* __restrict__ xz, const float* __restrict__ W_dt,
    const float* __restrict__ b_dt, const float* __restrict__ A_log,
    const float* __restrict__ D_param, float* __restrict__ ysum) {
  const int d = blockIdx.x * 256 + threadIdx.x;   // 0..1023
  const int b = blockIdx.y;                       // 0..63
  const float wdt = W_dt[d];
  const float bd  = b_dt[d];
  const float Dp  = D_param[d];
  float A[DSTATE], h[DSTATE];
#pragma unroll
  for (int n = 0; n < DSTATE; ++n) {
    A[n] = -__expf(A_log[(size_t)d * DSTATE + n]);
    h[n] = 0.0f;
  }
  float acc = 0.0f;
  const size_t rowbase = (size_t)b * S_;
  for (int s = 0; s < S_; ++s) {
    const size_t r = rowbase + s;
    const float4* xr = (const float4*)(xz + r * XZS);
    const float dtr = ((const float*)xr)[0];
    float Bt[16], Ct[16];
    *(float4*)&Bt[0] = xr[1]; *(float4*)&Bt[4] = xr[2];
    *(float4*)&Bt[8] = xr[3]; *(float4*)&Bt[12] = xr[4];
    *(float4*)&Ct[0] = xr[5]; *(float4*)&Ct[4] = xr[6];
    *(float4*)&Ct[8] = xr[7]; *(float4*)&Ct[12] = xr[8];
    const float uv = u[r * (size_t)DINNER + d];
    const float z  = fmaf(dtr, wdt, bd);
    const float dt = (z > 20.0f) ? z : log1pf(__expf(z));   // softplus
    const float du = dt * uv;
    float y = 0.0f;
#pragma unroll
    for (int n = 0; n < DSTATE; ++n) {
      const float dA = __expf(dt * A[n]);
      h[n] = fmaf(dA, h[n], du * Bt[n]);
      y    = fmaf(Ct[n], h[n], y);
    }
    y = fmaf(Dp, uv, y);
    if (x[r] != 0) acc += y;
  }
  ysum[(size_t)b * DINNER + d] = acc;
}

// ---------------------------------------------------------------------------
// Kernel 4: wcomb[i] = sum_dm W_head[dm] * W_out[dm, i]
// ---------------------------------------------------------------------------
__global__ __launch_bounds__(256) void wcomb_kernel(const float* __restrict__ W_out,
                                                    const float* __restrict__ W_head,
                                                    float* __restrict__ wc) {
  const int i = blockIdx.x * 256 + threadIdx.x;   // 0..1023
  float a = 0.0f;
  for (int dm = 0; dm < DMODEL; ++dm)
    a = fmaf(W_head[dm], W_out[(size_t)dm * DINNER + i], a);
  wc[i] = a;
}

// ---------------------------------------------------------------------------
// Kernel 5: out[b] = (sum_i ysum[b,i]*wcomb[i]) / max(count_b,1) + b_head
// ---------------------------------------------------------------------------
__global__ __launch_bounds__(256) void head_kernel(const int* __restrict__ x,
                                                   const float* __restrict__ ysum,
                                                   const float* __restrict__ wc,
                                                   const float* __restrict__ b_head,
                                                   float* __restrict__ out) {
  __shared__ float red[256];
  __shared__ float redc[256];
  const int b = blockIdx.x, tid = threadIdx.x;
  float dp = 0.0f, cnt = 0.0f;
  for (int i = tid; i < DINNER; i += 256)
    dp = fmaf(ysum[(size_t)b * DINNER + i], wc[i], dp);
  for (int s = tid; s < S_; s += 256)
    cnt += (x[(size_t)b * S_ + s] != 0) ? 1.0f : 0.0f;
  red[tid] = dp; redc[tid] = cnt;
  __syncthreads();
  for (int off = 128; off > 0; off >>= 1) {
    if (tid < off) { red[tid] += red[tid + off]; redc[tid] += redc[tid + off]; }
    __syncthreads();
  }
  if (tid == 0) out[b] = red[0] / fmaxf(redc[0], 1.0f) + b_head[0];
}

// ---------------------------------------------------------------------------
extern "C" void kernel_launch(void* const* d_in, const int* in_sizes, int n_in,
                              void* d_out, int out_size, void* d_ws, size_t ws_size,
                              hipStream_t stream) {
  (void)in_sizes; (void)n_in; (void)out_size; (void)ws_size;
  const int*   x       = (const int*)d_in[0];
  const float* tok_emb = (const float*)d_in[1];
  const float* W_in    = (const float*)d_in[2];
  const float* W_x     = (const float*)d_in[3];
  const float* W_dt    = (const float*)d_in[4];
  const float* b_dt    = (const float*)d_in[5];
  const float* A_log   = (const float*)d_in[6];
  const float* D_param = (const float*)d_in[7];
  const float* W_out   = (const float*)d_in[8];
  const float* W_head  = (const float*)d_in[9];
  const float* b_head  = (const float*)d_in[10];
  float* out = (float*)d_out;

  const size_t M = (size_t)B_ * S_;           // 131072
  float*  u      = (float*)d_ws;              // M * 1024 f32            (512 MB)
  float*  xz     = u + M * DINNER;            // M * 36 f32              ( 18 MB)
  float*  ysum   = xz + M * XZS;              // 64 * 1024 f32
  float*  wc     = ysum + (size_t)B_ * DINNER;// 1024 f32
  __bf16* temb16 = (__bf16*)(wc + 1024);      // 33278*512 bf16 (16B-aligned offset)
  __bf16* win16  = temb16 + (size_t)33278 * DMODEL;  // 1024*512 bf16

  // one-shot bf16 conversion of the GEMM operands
  const int n4_emb = 33278 * DMODEL / 4;      // 4259584
  const int n4_win = DINNER * DMODEL / 4;     // 131072
  cvt_bf16_kernel<<<(n4_emb + 255) / 256, 256, 0, stream>>>(tok_emb, (unsigned int*)temb16, n4_emb);
  cvt_bf16_kernel<<<(n4_win + 255) / 256, 256, 0, stream>>>(W_in, (unsigned int*)win16, n4_win);

  dim3 g1(1024, 8);
  gemm1_kernel<<<g1, 256, 0, stream>>>(x, temb16, win16, u);
  xz_kernel<<<16384, 256, 0, stream>>>(u, W_x, xz);
  dim3 gs(4, B_);
  scan_kernel<<<gs, 256, 0, stream>>>(x, u, xz, W_dt, b_dt, A_log, D_param, ysum);
  wcomb_kernel<<<4, 256, 0, stream>>>(W_out, W_head, wc);
  head_kernel<<<B_, 256, 0, stream>>>(x, ysum, wc, b_head, out);
}
